// NeighborAggregateLoss_27642409517745
// MI455X (gfx1250) — matrix operand order repaired and checked by reading, not compiled, not run on hardware
//
#include <hip/hip_runtime.h>
#include <hip/hip_bf16.h>

typedef __attribute__((ext_vector_type(16))) _Float16 v16h;
typedef __attribute__((ext_vector_type(8)))  _Float16 v8h;
typedef __attribute__((ext_vector_type(8)))  float    v8f;

#define N_ROWS 16384
#define DIM 64
#define TAU 0.28f
#define EPS 1e-8f
// exp(x/TAU) == exp2(x * log2(e)/TAU); v_exp_f32 is a base-2 exponential.
#define EXP_SCALE (1.44269504088896340736f / TAU)

// ---------------------------------------------------------------------------
// Kernel 1: row L2-normalize u and i, emit f16 copies of u_hat and
// w = u_hat + i_hat, plus pos_dot[m] = u_hat_m . i_hat_m.
// One wave32 per row (lane holds elements lane and lane+32).
// ---------------------------------------------------------------------------
__global__ __launch_bounds__(256)
void na_normalize_kernel(const float* __restrict__ U, const float* __restrict__ I,
                         _Float16* __restrict__ uh, _Float16* __restrict__ wh,
                         float* __restrict__ pos_dot) {
  int row  = (blockIdx.x * blockDim.x + threadIdx.x) >> 5;
  int lane = threadIdx.x & 31;
  if (row >= N_ROWS) return;

  const float* urow = U + (size_t)row * DIM;
  const float* irow = I + (size_t)row * DIM;
  float u0 = urow[lane], u1 = urow[lane + 32];
  float i0 = irow[lane], i1 = irow[lane + 32];

  float su = u0 * u0 + u1 * u1;
  float si = i0 * i0 + i1 * i1;
  #pragma unroll
  for (int s = 16; s > 0; s >>= 1) {
    su += __shfl_xor(su, s, 32);
    si += __shfl_xor(si, s, 32);
  }
  float du = fmaxf(sqrtf(su), 1e-12f);
  float di = fmaxf(sqrtf(si), 1e-12f);
  float un0 = u0 / du, un1 = u1 / du;
  float in0 = i0 / di, in1 = i1 / di;

  float pd = un0 * in0 + un1 * in1;
  #pragma unroll
  for (int s = 16; s > 0; s >>= 1) pd += __shfl_xor(pd, s, 32);

  _Float16* uo = uh + (size_t)row * DIM;
  _Float16* wo = wh + (size_t)row * DIM;
  uo[lane]      = (_Float16)un0;
  uo[lane + 32] = (_Float16)un1;
  wo[lane]      = (_Float16)(un0 + in0);
  wo[lane + 32] = (_Float16)(un1 + in1);
  if (lane == 0) pos_dot[row] = pd;
}

// ---------------------------------------------------------------------------
// Kernel 2: total[m] = sum_n exp( (u_hat @ w^T)[m][n] / TAU )
// One wave owns a 16-row M tile and sweeps all 16384 columns in 16-col tiles.
// Per tile: 2x v_wmma_f32_16x16x32_f16 (K=64), 8x v_exp_f32, per-lane row
// partials; final 16-lane shfl reduction. Uniform control flow -> EXEC all 1s.
// ---------------------------------------------------------------------------
__global__ __launch_bounds__(128)
void na_gemm_exp_rowsum_kernel(const _Float16* __restrict__ uh,
                               const _Float16* __restrict__ wh,
                               float* __restrict__ total) {
  int wave   = (blockIdx.x * blockDim.x + threadIdx.x) >> 5;  // 0..1023
  int lane   = threadIdx.x & 31;
  int half   = lane >> 4;     // 0: lanes 0-15, 1: lanes 16-31
  int l16    = lane & 15;
  int m_base = wave * 16;

  // A fragments: contiguous K order (matches B):
  //   lanes 0-15 : M=l16, K = 0..15
  //   lanes 16-31: M=l16, K = 16..31
  const _Float16* ap = uh + (size_t)(m_base + l16) * DIM + (half ? 16 : 0);
  v8h a00 = *(const v8h*)(ap);        // K k0+0..7
  v8h a01 = *(const v8h*)(ap + 8);    // K k0+8..15
  v8h a10 = *(const v8h*)(ap + 32);   // second K-half (+32)
  v8h a11 = *(const v8h*)(ap + 40);
  v16h a0 = __builtin_shufflevector(a00, a01, 0,1,2,3,4,5,6,7,8,9,10,11,12,13,14,15);
  v16h a1 = __builtin_shufflevector(a10, a11, 0,1,2,3,4,5,6,7,8,9,10,11,12,13,14,15);

  v8f acc = {0.f,0.f,0.f,0.f,0.f,0.f,0.f,0.f};  // per-lane row partial sums

  for (int nb = 0; nb < N_ROWS; nb += 16) {
    // B fragments: 16-bit B KxN layout: lane = column (l16), K split at lane 16.
    // Each lane reads 32 contiguous bytes of w[row].
    const _Float16* bp = wh + (size_t)(nb + l16) * DIM + (half ? 16 : 0);
    if (nb + 16 < N_ROWS)
      __builtin_prefetch(bp + 16 * DIM, 0, 3);   // global_prefetch_b8 next tile
    v16h b0 = *(const v16h*)(bp);        // K 0..15 / 16..31
    v16h b1 = *(const v16h*)(bp + 32);   // K 32..47 / 48..63

    v8f c = {0.f,0.f,0.f,0.f,0.f,0.f,0.f,0.f};
    c = __builtin_amdgcn_wmma_f32_16x16x32_f16(false, a0, false, b0,
                                               (short)0, c, false, false);
    c = __builtin_amdgcn_wmma_f32_16x16x32_f16(false, a1, false, b1,
                                               (short)0, c, false, false);
    #pragma unroll
    for (int r = 0; r < 8; ++r)
      acc[r] += __builtin_amdgcn_exp2f(c[r] * EXP_SCALE);
  }

  // C layout: VGPR r holds (row m_base+r, col=lane) for lanes 0-15 and
  // (row m_base+8+r, col=lane-16) for lanes 16-31. Reduce within 16-lane groups.
  #pragma unroll
  for (int r = 0; r < 8; ++r) {
    float v = acc[r];
    v += __shfl_xor(v, 1, 32);
    v += __shfl_xor(v, 2, 32);
    v += __shfl_xor(v, 4, 32);
    v += __shfl_xor(v, 8, 32);
    acc[r] = v;
  }
  if (l16 == 0) {
    int row0 = m_base + (half ? 8 : 0);
    #pragma unroll
    for (int r = 0; r < 8; ++r) total[row0 + r] = acc[r];
  }
}

// ---------------------------------------------------------------------------
// Kernel 3: loss = mean( log(total + EPS) - pos_dot / TAU )
// ---------------------------------------------------------------------------
__global__ __launch_bounds__(1024)
void na_loss_reduce_kernel(const float* __restrict__ total,
                           const float* __restrict__ pos_dot,
                           float* __restrict__ out) {
  __shared__ float red[32];
  int tid = threadIdx.x;
  float s = 0.f;
  for (int m = tid; m < N_ROWS; m += blockDim.x)
    s += logf(total[m] + EPS) - pos_dot[m] * (1.0f / TAU);

  #pragma unroll
  for (int k = 16; k > 0; k >>= 1) s += __shfl_xor(s, k, 32);
  int wid = tid >> 5, lane = tid & 31;
  if (lane == 0) red[wid] = s;
  __syncthreads();
  if (wid == 0) {
    float v = (lane < (int)(blockDim.x >> 5)) ? red[lane] : 0.f;
    #pragma unroll
    for (int k = 16; k > 0; k >>= 1) v += __shfl_xor(v, k, 32);
    if (lane == 0) out[0] = v / (float)N_ROWS;
  }
}

// ---------------------------------------------------------------------------
extern "C" void kernel_launch(void* const* d_in, const int* in_sizes, int n_in,
                              void* d_out, int out_size, void* d_ws, size_t ws_size,
                              hipStream_t stream) {
  const float* U = (const float*)d_in[0];
  const float* I = (const float*)d_in[1];

  char* ws = (char*)d_ws;
  size_t half_bytes = (size_t)N_ROWS * DIM * sizeof(_Float16);  // 2 MB each
  _Float16* uh      = (_Float16*)(ws);
  _Float16* wh      = (_Float16*)(ws + half_bytes);
  float*    pos_dot = (float*)   (ws + 2 * half_bytes);
  float*    total   = (float*)   (ws + 2 * half_bytes + (size_t)N_ROWS * sizeof(float));

  // 1 wave per row, 8 rows per 256-thread block.
  na_normalize_kernel<<<N_ROWS / 8, 256, 0, stream>>>(U, I, uh, wh, pos_dot);
  // 1024 M-tiles, 4 waves per 128-thread block -> 256 blocks.
  na_gemm_exp_rowsum_kernel<<<(N_ROWS / 16) / 4, 128, 0, stream>>>(uh, wh, total);
  na_loss_reduce_kernel<<<1, 1024, 0, stream>>>(total, pos_dot, (float*)d_out);
}
